// EmbAGNNRecluster_69157563400721
// MI455X (gfx1250) — matrix operand order repaired and checked by reading, not compiled, not run on hardware
//
#include <hip/hip_runtime.h>
#include <hip/hip_bf16.h>

#define EPSF 1e-5f

// gfx1250 has a hardware V_TANH_F32 transcendental; use it when exposed.
#if __has_builtin(__builtin_amdgcn_tanhf)
__device__ __forceinline__ float fast_tanh(float x) { return __builtin_amdgcn_tanhf(x); }
#elif __has_builtin(__builtin_amdgcn_tanh_f32)
__device__ __forceinline__ float fast_tanh(float x) { return __builtin_amdgcn_tanh_f32(x); }
#else
__device__ __forceinline__ float fast_tanh(float x) { return tanhf(x); }
#endif

typedef __attribute__((ext_vector_type(16))) _Float16 v16h;
typedef __attribute__((ext_vector_type(8)))  float    v8f;

// ---------------------------------------------------------------------------
// Small helpers
// ---------------------------------------------------------------------------
__device__ __forceinline__ v16h hzero() {
  v16h a;
#pragma unroll
  for (int i = 0; i < 16; ++i) a[i] = (_Float16)0.0f;
  return a;
}

__device__ __forceinline__ v8f fzero() {
  v8f c;
#pragma unroll
  for (int i = 0; i < 8; ++i) c[i] = 0.0f;
  return c;
}

__device__ __forceinline__ v8f wmma16(v16h a, v16h b, v8f c) {
  // D = A(16x32 f16) * B(32x16 f16) + C(16x16 f32)
  return __builtin_amdgcn_wmma_f32_16x16x32_f16(false, a, false, b, (short)0, c,
                                                false, false);
}

// Wave-local compiler fence: same-wave LDS ops are in-order at the LDS; we
// only need to stop compiler reordering across the cross-lane LDS hand-off.
__device__ __forceinline__ void wfence() {
  asm volatile("" ::: "memory");
  __builtin_amdgcn_wave_barrier();
  asm volatile("" ::: "memory");
}

// Build a WMMA B operand from a (rows x 8) row-major f32 weight matrix,
// zero-padded to 32x16.  B layout (f16, 32x16): lane half 'hi' holds
// K = k + 8*hi (k < 8) or 16 + 8*hi + (k - 8) (k >= 8), N = lane%16.
__device__ __forceinline__ v16h make_B(const float* W, int rows, int n, int hi) {
  v16h b = hzero();
  if (n < 8) {
#pragma unroll
    for (int k = 0; k < 16; ++k) {
      int K = (k < 8) ? (k + 8 * hi) : (16 + 8 * hi + (k - 8));
      if (K < rows) b[k] = (_Float16)W[K * 8 + n];
    }
  }
  return b;
}

// C-layout z (raw WMMA accumulator, 8 valid columns) -> LDS -> A-layout, with
// bias + LayerNorm + tanh computed IN-LANE on the A side (a lo-lane holds all
// 8 features of its row contiguously) -> no cross-lane shuffles at all.
// bz/gz/ez are wave-uniform 8-element arrays preloaded in registers.
__device__ __forceinline__ v16h relayout_ln_tanh(float* T, v8f c,
                                                 const float* bz, const float* gz,
                                                 const float* ez, int n, int hi) {
  if (n < 8) {
#pragma unroll
    for (int v = 0; v < 8; ++v) T[(v + 8 * hi) * 8 + n] = c[v];
  }
  wfence();
  v16h A = hzero();
  if (hi == 0) {
    const float4* row = (const float4*)(T + n * 8);
    const float4 r0 = row[0];
    const float4 r1 = row[1];
    float z[8] = {r0.x, r0.y, r0.z, r0.w, r1.x, r1.y, r1.z, r1.w};
    float m = 0.f;
#pragma unroll
    for (int k = 0; k < 8; ++k) { z[k] += bz[k]; m += z[k]; }
    m *= 0.125f;
    float var = 0.f;
#pragma unroll
    for (int k = 0; k < 8; ++k) { const float d = z[k] - m; var += d * d; }
    var *= 0.125f;
    const float r = rsqrtf(var + EPSF);
#pragma unroll
    for (int k = 0; k < 8; ++k)
      A[k] = (_Float16)fast_tanh((z[k] - m) * r * gz[k] + ez[k]);
  }
  wfence();
  return A;
}

// ---------------------------------------------------------------------------
// Edge net: per 16-edge tile, MLP 16->8->8->8->1; optionally sigmoid+scatter.
// ---------------------------------------------------------------------------
struct EdgeW {
  const float *W1, *b1, *g1, *be1;
  const float *W2, *b2, *g2, *be2;
  const float *W3, *b3, *g3, *be3;
  const float *Wf, *bf;
};

template <bool SCATTER>
__global__ __launch_bounds__(256) void edge_net_kernel(
    const float* __restrict__ feat, const int* __restrict__ ei, int E_,
    float* __restrict__ mi, float* __restrict__ mo,
    float* __restrict__ scores, EdgeW p) {
  __shared__ float lds[8][144];  // per-wave: 16x8 tile + 16 score slots
  const int lane = threadIdx.x & 31;
  const int n    = lane & 15;
  const int hi   = lane >> 4;
  float* T = lds[threadIdx.x >> 5];
  float* S = T + 128;

  // Weight operands in WMMA B-layout (loaded once per wave, reused per tile).
  v16h B1 = make_B(p.W1, 16, n, hi);
  v16h B2 = make_B(p.W2, 8, n, hi);
  v16h B3 = make_B(p.W3, 8, n, hi);
  v16h Bf = hzero();
  if (lane == 0) {
#pragma unroll
    for (int k = 0; k < 8; ++k) Bf[k] = (_Float16)p.Wf[k];  // 8x1 in col 0
  }
  // Wave-uniform LN/bias params preloaded (scalar loads, hoisted).
  float b1a[8], g1a[8], e1a[8], b2a[8], g2a[8], e2a[8], b3a[8], g3a[8], e3a[8];
#pragma unroll
  for (int k = 0; k < 8; ++k) {
    b1a[k] = p.b1[k]; g1a[k] = p.g1[k]; e1a[k] = p.be1[k];
    b2a[k] = p.b2[k]; g2a[k] = p.g2[k]; e2a[k] = p.be2[k];
    b3a[k] = p.b3[k]; g3a[k] = p.g3[k]; e3a[k] = p.be3[k];
  }
  const float bfv = p.bf[0];

  const int wid    = blockIdx.x * (blockDim.x >> 5) + (threadIdx.x >> 5);
  const int nwaves = gridDim.x * (blockDim.x >> 5);
  const int ntiles = (E_ + 15) >> 4;

  for (int t = wid; t < ntiles; t += nwaves) {
    const int  me = (t << 4) + n;       // edge handled by this lane (row M=n)
    const bool ev = me < E_;
    const int  mc = ev ? me : (E_ - 1);
    const int  src = ei[mc];
    const int  dst = ei[E_ + mc];
    const int  node = hi ? dst : src;   // lo: feat[src] (K 0..7), hi: feat[dst] (K 8..15)
    const float* fp = feat + (size_t)node * 8;
    float f[8];
#pragma unroll
    for (int i = 0; i < 8; ++i) f[i] = fp[i];

    v16h A = hzero();
#pragma unroll
    for (int i = 0; i < 8; ++i) A[i] = (_Float16)f[i];

    v8f  c  = wmma16(A, B1, fzero());
    v16h A2 = relayout_ln_tanh(T, c, b1a, g1a, e1a, n, hi);
    c       = wmma16(A2, B2, fzero());
    v16h A3 = relayout_ln_tanh(T, c, b2a, g2a, e2a, n, hi);
    c       = wmma16(A3, B3, fzero());
    v16h A4 = relayout_ln_tanh(T, c, b3a, g3a, e3a, n, hi);
    c       = wmma16(A4, Bf, fzero());  // score for row v+8*hi in col 0 (n==0)

    if (n == 0) {
#pragma unroll
      for (int v = 0; v < 8; ++v) S[v + 8 * hi] = c[v] + bfv;
    }
    wfence();
    const float sc = S[n];  // score of this lane's edge

    if (SCATTER) {
      const float ewt = 1.0f / (1.0f + __expf(-sc));
      if (ev) {
        // lo lane holds feat[src] -> mi[dst] += e*feat[src]
        // hi lane holds feat[dst] -> mo[src] += e*feat[dst]
        float* dp = hi ? (mo + (size_t)src * 8) : (mi + (size_t)dst * 8);
#pragma unroll
        for (int i = 0; i < 8; ++i)
          (void)__hip_atomic_fetch_add(dp + i, ewt * f[i], __ATOMIC_RELAXED,
                                       __HIP_MEMORY_SCOPE_AGENT);
      }
    } else {
      if (!hi && ev) scores[me] = sc;  // raw score (no sigmoid), matches ref
    }
  }
}

// ---------------------------------------------------------------------------
// Node net: per 16-node tile, MLP 24->8->8->8->8 with residual add.
// ---------------------------------------------------------------------------
struct NodeW {
  const float *W1, *b1, *g1, *be1;
  const float *W2, *b2, *g2, *be2;
  const float *W3, *b3, *g3, *be3;
  const float *Wf, *bf;
};

__global__ __launch_bounds__(256) void node_net_kernel(
    const float* __restrict__ feat, const float* __restrict__ mi,
    const float* __restrict__ mo, float* __restrict__ outf, int N_, NodeW p) {
  __shared__ float lds[8][128];
  const int lane = threadIdx.x & 31;
  const int n    = lane & 15;
  const int hi   = lane >> 4;
  const int n7   = n & 7;
  float* T = lds[threadIdx.x >> 5];

  v16h B1 = make_B(p.W1, 24, n, hi);
  v16h B2 = make_B(p.W2, 8, n, hi);
  v16h B3 = make_B(p.W3, 8, n, hi);
  v16h Bf = make_B(p.Wf, 8, n, hi);
  float b1a[8], g1a[8], e1a[8], b2a[8], g2a[8], e2a[8], b3a[8], g3a[8], e3a[8];
#pragma unroll
  for (int k = 0; k < 8; ++k) {
    b1a[k] = p.b1[k]; g1a[k] = p.g1[k]; e1a[k] = p.be1[k];
    b2a[k] = p.b2[k]; g2a[k] = p.g2[k]; e2a[k] = p.be2[k];
    b3a[k] = p.b3[k]; g3a[k] = p.g3[k]; e3a[k] = p.be3[k];
  }
  const float bfv = p.bf[n7];  // per-column final bias

  const int wid    = blockIdx.x * (blockDim.x >> 5) + (threadIdx.x >> 5);
  const int nwaves = gridDim.x * (blockDim.x >> 5);
  const int ntiles = (N_ + 15) >> 4;

  for (int t = wid; t < ntiles; t += nwaves) {
    const int  nd = (t << 4) + n;
    const bool nv = nd < N_;
    const int  nc = nv ? nd : (N_ - 1);
    // Input [mi | mo | feat], K=24 padded to 32:
    //   lo: K0..7 = mi,  K16..23 = feat ; hi: K8..15 = mo, K24..31 = 0
    const float* p0 = (hi ? mo : mi) + (size_t)nc * 8;
    const float* pf = feat + (size_t)nc * 8;
    v16h A = hzero();
#pragma unroll
    for (int i = 0; i < 8; ++i) A[i] = (_Float16)p0[i];
    if (hi == 0) {
#pragma unroll
      for (int i = 0; i < 8; ++i) A[8 + i] = (_Float16)pf[i];
    }

    v8f  c  = wmma16(A, B1, fzero());
    v16h A2 = relayout_ln_tanh(T, c, b1a, g1a, e1a, n, hi);
    c       = wmma16(A2, B2, fzero());
    v16h A3 = relayout_ln_tanh(T, c, b2a, g2a, e2a, n, hi);
    c       = wmma16(A3, B3, fzero());
    v16h A4 = relayout_ln_tanh(T, c, b3a, g3a, e3a, n, hi);
    c       = wmma16(A4, Bf, fzero());

    // Residual + store: element (M = v+8*hi, col n), n<8 valid
    if (n < 8) {
#pragma unroll
      for (int v = 0; v < 8; ++v) {
        const int node2 = (t << 4) + v + 8 * hi;
        if (node2 < N_) {
          const size_t o = (size_t)node2 * 8 + n;
          outf[o] = c[v] + bfv + feat[o];
        }
      }
    }
  }
}

// ---------------------------------------------------------------------------
// Embedding MLPs (3/19 -> 32 -> 32 -> 32 -> 32 -> 8), per-thread scalar VALU
// (wave-uniform weights hit the scalar/L2 caches; ~1.5 GMAC total -> cheap).
// ---------------------------------------------------------------------------
struct EmbW {
  const float *W0, *b0, *W1, *b1, *W2, *b2, *W3, *b3, *We, *be;
};

__device__ __forceinline__ void hid32x32(const float* __restrict__ W,
                                         const float* __restrict__ b, float* h) {
  float h2[32];
#pragma unroll 4
  for (int o = 0; o < 32; ++o) {
    float s = b[o];
#pragma unroll
    for (int i = 0; i < 32; ++i) s += h[i] * W[i * 32 + o];
    h2[o] = fast_tanh(s);
  }
#pragma unroll
  for (int o = 0; o < 32; ++o) h[o] = h2[o];
}

template <int IN>
__device__ __forceinline__ void run_emb(const float* in, const EmbW& p, float* out8) {
  float h[32];
#pragma unroll 4
  for (int o = 0; o < 32; ++o) {
    float s = p.b0[o];
#pragma unroll
    for (int i = 0; i < IN; ++i) s += in[i] * p.W0[i * 32 + o];
    h[o] = fast_tanh(s);
  }
  hid32x32(p.W1, p.b1, h);
  hid32x32(p.W2, p.b2, h);
  hid32x32(p.W3, p.b3, h);
#pragma unroll
  for (int o = 0; o < 8; ++o) {
    float s = p.be[o];
#pragma unroll
    for (int i = 0; i < 32; ++i) s += h[i] * p.We[i * 8 + o];
    out8[o] = s;
  }
}

__global__ void emb1_infeat_kernel(const float* __restrict__ x, EmbW p,
                                   const float* __restrict__ iW,
                                   const float* __restrict__ ib,
                                   const float* __restrict__ ig,
                                   const float* __restrict__ ibe,
                                   float* __restrict__ spatial,
                                   float* __restrict__ feat, int N_) {
  const int t = blockIdx.x * blockDim.x + threadIdx.x;
  if (t >= N_) return;
  float in3[3];
#pragma unroll
  for (int i = 0; i < 3; ++i) in3[i] = x[(size_t)t * 3 + i];
  float sp[8];
  run_emb<3>(in3, p, sp);
#pragma unroll
  for (int o = 0; o < 8; ++o) spatial[(size_t)t * 8 + o] = sp[o];
  // infeat: [spatial(8), x(3)] @ W(11x8) + b -> LN -> tanh
  float z[8];
#pragma unroll
  for (int o = 0; o < 8; ++o) {
    float s = ib[o];
#pragma unroll
    for (int i = 0; i < 8; ++i) s += sp[i] * iW[i * 8 + o];
#pragma unroll
    for (int i = 0; i < 3; ++i) s += in3[i] * iW[(8 + i) * 8 + o];
    z[o] = s;
  }
  float m = 0.f;
#pragma unroll
  for (int o = 0; o < 8; ++o) m += z[o];
  m *= 0.125f;
  float var = 0.f;
#pragma unroll
  for (int o = 0; o < 8; ++o) { float d = z[o] - m; var += d * d; }
  var *= 0.125f;
  const float r = rsqrtf(var + EPSF);
#pragma unroll
  for (int o = 0; o < 8; ++o)
    feat[(size_t)t * 8 + o] = fast_tanh((z[o] - m) * r * ig[o] + ibe[o]);
}

__global__ void emb2_kernel(const float* __restrict__ sp1,
                            const float* __restrict__ x,
                            const float* __restrict__ ft,
                            float* __restrict__ outsp, int N_, EmbW p) {
  const int t = blockIdx.x * blockDim.x + threadIdx.x;
  if (t >= N_) return;
  float in[19];
#pragma unroll
  for (int i = 0; i < 8; ++i) in[i] = sp1[(size_t)t * 8 + i];
#pragma unroll
  for (int i = 0; i < 3; ++i) in[8 + i] = x[(size_t)t * 3 + i];
#pragma unroll
  for (int i = 0; i < 8; ++i) in[11 + i] = ft[(size_t)t * 8 + i];
  float o8[8];
  run_emb<19>(in, p, o8);
#pragma unroll
  for (int i = 0; i < 8; ++i) outsp[(size_t)t * 8 + i] = o8[i];
}

__global__ void zero_kernel(float4* __restrict__ ptr, long long cnt4) {
  long long i = (long long)blockIdx.x * blockDim.x + threadIdx.x;
  const long long stride = (long long)gridDim.x * blockDim.x;
  const float4 z = {0.f, 0.f, 0.f, 0.f};
  for (; i < cnt4; i += stride) ptr[i] = z;
}

// ---------------------------------------------------------------------------
// Orchestration
// ---------------------------------------------------------------------------
extern "C" void kernel_launch(void* const* d_in, const int* in_sizes, int n_in,
                              void* d_out, int out_size, void* d_ws, size_t ws_size,
                              hipStream_t stream) {
  (void)n_in; (void)out_size; (void)ws_size;
  const float* x  = (const float*)d_in[0];
  const int* ei1  = (const int*)d_in[1];
  const int* ei2  = (const int*)d_in[2];
  const int N_ = in_sizes[0] / 3;
  const int E_ = in_sizes[1] / 2;

  auto F = [&](int i) { return (const float*)d_in[i]; };
  // params pytree (sorted dict keys: edge, emb1, emb2, infeat, node;
  // within mlp_ln: final, hidden; within emb: emb, layers)
  EdgeW ew  = { F(5),  F(6),  F(7),  F(8),
                F(9),  F(10), F(11), F(12),
                F(13), F(14), F(15), F(16),
                F(3),  F(4) };
  EmbW  e1  = { F(19), F(20), F(21), F(22), F(23), F(24), F(25), F(26), F(17), F(18) };
  EmbW  e2  = { F(29), F(30), F(31), F(32), F(33), F(34), F(35), F(36), F(27), F(28) };
  const float *iW = F(37), *ib = F(38), *ig = F(39), *ibe = F(40);
  NodeW nw  = { F(43), F(44), F(45), F(46),
                F(47), F(48), F(49), F(50),
                F(51), F(52), F(53), F(54),
                F(41), F(42) };

  float* outScores  = (float*)d_out;       // E
  float* outSpatial = outScores + E_;      // N*8

  const size_t N8 = (size_t)N_ * 8;
  float* spatial1 = (float*)d_ws;          // N*8
  float* featA    = spatial1 + N8;         // N*8
  float* featB    = featA + N8;            // N*8
  float* mi       = featB + N8;            // N*8
  float* mo       = mi + N8;               // N*8  (mi,mo contiguous)

  const int TB = 256;
  const int nodeBlocks = (N_ + TB - 1) / TB;

  // 1) emb1 + infeat -> spatial1, featA
  emb1_infeat_kernel<<<nodeBlocks, TB, 0, stream>>>(x, e1, iW, ib, ig, ibe,
                                                    spatial1, featA, N_);
  // 2) iteration 1 on edge_index1
  zero_kernel<<<1024, TB, 0, stream>>>((float4*)mi, (long long)(N8 * 2 / 4));
  edge_net_kernel<true><<<4096, TB, 0, stream>>>(featA, ei1, E_, mi, mo, nullptr, ew);
  node_net_kernel<<<1600, TB, 0, stream>>>(featA, mi, mo, featB, N_, nw);
  // 3) emb2 -> output spatial
  emb2_kernel<<<nodeBlocks, TB, 0, stream>>>(spatial1, x, featB, outSpatial, N_, e2);
  // 4) iteration 2 on edge_index2
  zero_kernel<<<1024, TB, 0, stream>>>((float4*)mi, (long long)(N8 * 2 / 4));
  edge_net_kernel<true><<<4096, TB, 0, stream>>>(featB, ei2, E_, mi, mo, nullptr, ew);
  node_net_kernel<<<1600, TB, 0, stream>>>(featB, mi, mo, featA, N_, nw);
  // 5) final scores (raw, no sigmoid) on edge_index2
  edge_net_kernel<false><<<4096, TB, 0, stream>>>(featA, ei2, E_, nullptr, nullptr,
                                                  outScores, ew);
}
